// MultiHeadAttention_54090818126000
// MI455X (gfx1250) — matrix operand order repaired and checked by reading, not compiled, not run on hardware
//
#include <hip/hip_runtime.h>
#include <hip/hip_bf16.h>

#define EMBED 1024
#define HEADS 16
#define DEPTH 64
#define BATCH 2
#define SEQ   2048

typedef __attribute__((ext_vector_type(16))) __bf16 v16bf;
typedef __attribute__((ext_vector_type(8)))  float  v8f;

union BF16x16 {
    v16bf v;
    unsigned short u[16];
};

__device__ __forceinline__ unsigned short bf_bits(__bf16 h) {
    union { __bf16 h; unsigned short u; } c;
    c.h = h;
    return c.u;
}

__device__ __forceinline__ v8f wmma_bf16(v16bf a, v16bf b, v8f c) {
    // (neg_a, A, neg_b, B, c_mod, C, reuse_a, reuse_b)
    return __builtin_amdgcn_wmma_f32_16x16x32_bf16(false, a, false, b,
                                                   (short)0, c, false, false);
}

// ---- Fragment loaders (contiguous k = 16*(lane/16) + s, same order in A and B, wave32) ----

// A: 16x32 (MxK), source fp32 row-major, leading dim lda.
// lane<16: row M=lane, K = 0..15; lane>=16: same rows, K = 16..31
__device__ __forceinline__ v16bf load_a_f32(const float* __restrict__ src,
                                            int m0, int k0, int lda, int lane) {
    int r  = lane & 15;
    int kh = (lane >> 4) << 4;            // 0 or 16
    const float* p = src + (m0 + r) * lda + k0;
    v16bf A;
#pragma unroll
    for (int j = 0; j < 8; ++j) {
        A[j]     = (__bf16)p[kh + j];          // lowers to v_cvt_pk_bf16_f32
        A[j + 8] = (__bf16)p[kh + j + 8];
    }
    return A;
}

__device__ __forceinline__ v16bf load_a_bf16(const unsigned short* __restrict__ src,
                                             int m0, int k0, int lda, int lane) {
    int r  = lane & 15;
    int kh = (lane >> 4) << 4;
    const unsigned short* p = src + (m0 + r) * lda + k0;
    BF16x16 A;
#pragma unroll
    for (int j = 0; j < 8; ++j) {
        A.u[j]     = p[kh + j];
        A.u[j + 8] = p[kh + j + 8];
    }
    return A.v;
}

// B: 32x16 (KxN) where B[k][n] = srcT[n][k], srcT row-major [N,K] (weight rows).
// lane<16: column n=lane, K=0..15; lane>=16: column n=lane-16, K=16..31
__device__ __forceinline__ v16bf load_bt_f32(const float* __restrict__ srcT,
                                             int n0, int k0, int ldk, int lane) {
    int c  = lane & 15;
    int kh = (lane >> 4) << 4;            // 0 or 16
    const float* p = srcT + (n0 + c) * ldk + k0 + kh;
    v16bf B;
#pragma unroll
    for (int j = 0; j < 16; ++j) B[j] = (__bf16)p[j];
    return B;
}

__device__ __forceinline__ v16bf load_bt_bf16(const unsigned short* __restrict__ srcT,
                                              int n0, int k0, int ldk, int lane) {
    int c  = lane & 15;
    int kh = (lane >> 4) << 4;
    const unsigned short* p = srcT + (n0 + c) * ldk + k0 + kh;
    BF16x16 B;
#pragma unroll
    for (int j = 0; j < 16; ++j) B.u[j] = p[j];
    return B.v;
}

// B: 32x16 (KxN), source row-major [K,N] bf16 (V matrix), leading dim ldn
__device__ __forceinline__ v16bf load_b_bf16(const unsigned short* __restrict__ src,
                                             int k0, int n0, int ldn, int lane) {
    int c  = lane & 15;
    int kh = (lane >> 4) << 4;
    const unsigned short* p = src + (k0 + kh) * ldn + n0 + c;
    BF16x16 B;
#pragma unroll
    for (int j = 0; j < 16; ++j) B.u[j] = p[j * ldn];
    return B.v;
}

// =============================================================
// Kernel 1: QKV projection. C[m,n] = x[m,:] . W[n,:] + b[n]
// wave tile: 16(M) x 64(N), A fragment reused across 4 WMMAs.
// writes bf16 in [B, H, S, DEPTH] layout. blockIdx.z picks Q/K/V.
// =============================================================
__global__ __launch_bounds__(256) void qkv_kernel(
    const float* __restrict__ x,
    const float* __restrict__ wq, const float* __restrict__ bq,
    const float* __restrict__ wk, const float* __restrict__ bk,
    const float* __restrict__ wv, const float* __restrict__ bv,
    unsigned short* __restrict__ qout,
    unsigned short* __restrict__ kout,
    unsigned short* __restrict__ vout) {
    const float* w; const float* bias; unsigned short* out;
    if (blockIdx.z == 0)      { w = wq; bias = bq; out = qout; }
    else if (blockIdx.z == 1) { w = wk; bias = bk; out = kout; }
    else                      { w = wv; bias = bv; out = vout; }

    int lane = threadIdx.x & 31;
    int wave = threadIdx.x >> 5;
    int n0 = blockIdx.x * 512 + wave * 64;   // 8 waves x 64 cols = 512
    int m0 = blockIdx.y * 16;

    v8f acc[4] = {};
    for (int k0 = 0; k0 < EMBED; k0 += 32) {
        if (k0 + 32 < EMBED) {
            __builtin_prefetch(x + (m0 + (lane & 15)) * EMBED + k0 + 32, 0, 3);
            __builtin_prefetch(w + (n0 + (lane & 15)) * EMBED + k0 + 32, 0, 3);
        }
        v16bf a = load_a_f32(x, m0, k0, EMBED, lane);
#pragma unroll
        for (int t = 0; t < 4; ++t) {
            v16bf b = load_bt_f32(w, n0 + t * 16, k0, EMBED, lane);
            acc[t] = wmma_bf16(a, b, acc[t]);
        }
    }

    int mh = (lane >> 4) << 3;
#pragma unroll
    for (int t = 0; t < 4; ++t) {
        int n = n0 + t * 16 + (lane & 15);
        float bval = bias[n];
        int h = n >> 6, dd = n & 63;
#pragma unroll
        for (int i = 0; i < 8; ++i) {
            int m = m0 + mh + i;
            int bb = m >> 11, s = m & (SEQ - 1);
            out[(((size_t)(bb * HEADS + h) * SEQ) + s) * DEPTH + dd] =
                bf_bits((__bf16)(acc[t][i] + bval));
        }
    }
}

// =============================================================
// Kernel 2: logits[bh,q,k] = (Q[q,:].K[k,:]) / 8 + mask[b,k]*(-1e9)
// wave tile: 16(q) x 64(keys); K-dim (=64) fully unrolled.
// =============================================================
__global__ __launch_bounds__(256) void logits_kernel(
    const unsigned short* __restrict__ qb,
    const unsigned short* __restrict__ kb,
    const int* __restrict__ mask,
    float* __restrict__ attn) {
    int bh = blockIdx.z;
    int b  = bh >> 4;
    int lane = threadIdx.x & 31;
    int wave = threadIdx.x >> 5;
    int kcol0 = blockIdx.x * 512 + wave * 64;
    int q0    = blockIdx.y * 16;

    const unsigned short* Q  = qb + (size_t)bh * SEQ * DEPTH;
    const unsigned short* Kh = kb + (size_t)bh * SEQ * DEPTH;

    v8f acc[4] = {};
#pragma unroll
    for (int k0 = 0; k0 < DEPTH; k0 += 32) {
        v16bf a = load_a_bf16(Q, q0, k0, DEPTH, lane);
#pragma unroll
        for (int t = 0; t < 4; ++t) {
            v16bf bm = load_bt_bf16(Kh, kcol0 + t * 16, k0, DEPTH, lane);
            acc[t] = wmma_bf16(a, bm, acc[t]);
        }
    }

    int mh = (lane >> 4) << 3;
    float* dst = attn + (size_t)bh * SEQ * SEQ;
#pragma unroll
    for (int t = 0; t < 4; ++t) {
        int kcol = kcol0 + t * 16 + (lane & 15);
        float mval = (float)mask[b * SEQ + kcol] * (-1e9f);
#pragma unroll
        for (int i = 0; i < 8; ++i) {
            int q = q0 + mh + i;
            dst[(size_t)q * SEQ + kcol] = acc[t][i] * 0.125f + mval;
        }
    }
}

// =============================================================
// Kernel 3: in-place row softmax over 2048 keys, 1 block / row
// =============================================================
__global__ __launch_bounds__(256) void softmax_kernel(float* __restrict__ attn) {
    __shared__ float red[256];
    float* p = attn + (size_t)blockIdx.x * SEQ;
    int t = threadIdx.x;
    float vals[8];
    float m = -3.4e38f;
#pragma unroll
    for (int i = 0; i < 8; ++i) { vals[i] = p[t + i * 256]; m = fmaxf(m, vals[i]); }
    red[t] = m; __syncthreads();
    for (int s = 128; s > 0; s >>= 1) {
        if (t < s) red[t] = fmaxf(red[t], red[t + s]);
        __syncthreads();
    }
    m = red[0]; __syncthreads();
    float sum = 0.f;
#pragma unroll
    for (int i = 0; i < 8; ++i) { vals[i] = __expf(vals[i] - m); sum += vals[i]; }
    red[t] = sum; __syncthreads();
    for (int s = 128; s > 0; s >>= 1) {
        if (t < s) red[t] += red[t + s];
        __syncthreads();
    }
    float inv = 1.0f / red[0];
#pragma unroll
    for (int i = 0; i < 8; ++i) p[t + i * 256] = vals[i] * inv;
}

// =============================================================
// Kernel 4: ctx = attn @ V. Wave computes full 16(q) x 64(d) block,
// attn fragment reused across all 4 d-subtiles. fp32 out, [B,S,D].
// =============================================================
__global__ __launch_bounds__(256) void ctx_kernel(
    const float* __restrict__ attn,
    const unsigned short* __restrict__ vb,
    float* __restrict__ ctx) {
    int bh = blockIdx.x;
    int b = bh >> 4, h = bh & 15;
    int lane = threadIdx.x & 31;
    int wave = threadIdx.x >> 5;
    int q0 = blockIdx.y * 128 + wave * 16;   // 8 waves x 16 q-rows

    const float* A = attn + (size_t)bh * SEQ * SEQ;
    const unsigned short* V = vb + (size_t)bh * SEQ * DEPTH;

    v8f acc[4] = {};
    for (int k0 = 0; k0 < SEQ; k0 += 32) {
        if (k0 + 32 < SEQ) {
            __builtin_prefetch(A + (size_t)(q0 + (lane & 15)) * SEQ + k0 + 32, 0, 3);
        }
        v16bf a = load_a_f32(A, q0, k0, SEQ, lane);
#pragma unroll
        for (int t = 0; t < 4; ++t) {
            v16bf bm = load_b_bf16(V, k0, t * 16, DEPTH, lane);
            acc[t] = wmma_bf16(a, bm, acc[t]);
        }
    }

    int mh = (lane >> 4) << 3;
#pragma unroll
    for (int t = 0; t < 4; ++t) {
        int dd = t * 16 + (lane & 15);
#pragma unroll
        for (int i = 0; i < 8; ++i) {
            int q = q0 + mh + i;
            ctx[((size_t)(b * SEQ + q)) * EMBED + h * DEPTH + dd] = acc[t][i];
        }
    }
}

// =============================================================
// Kernel 5: out = ctx @ dense_w^T + dense_b  (fp32 out)
// wave tile: 16(M) x 64(N)
// =============================================================
__global__ __launch_bounds__(256) void dense_kernel(
    const float* __restrict__ ctx,
    const float* __restrict__ w,
    const float* __restrict__ bias,
    float* __restrict__ out) {
    int lane = threadIdx.x & 31;
    int wave = threadIdx.x >> 5;
    int n0 = blockIdx.x * 512 + wave * 64;
    int m0 = blockIdx.y * 16;

    v8f acc[4] = {};
    for (int k0 = 0; k0 < EMBED; k0 += 32) {
        if (k0 + 32 < EMBED) {
            __builtin_prefetch(ctx + (m0 + (lane & 15)) * EMBED + k0 + 32, 0, 3);
            __builtin_prefetch(w + (n0 + (lane & 15)) * EMBED + k0 + 32, 0, 3);
        }
        v16bf a = load_a_f32(ctx, m0, k0, EMBED, lane);
#pragma unroll
        for (int t = 0; t < 4; ++t) {
            v16bf bm = load_bt_f32(w, n0 + t * 16, k0, EMBED, lane);
            acc[t] = wmma_bf16(a, bm, acc[t]);
        }
    }

    int mh = (lane >> 4) << 3;
#pragma unroll
    for (int t = 0; t < 4; ++t) {
        int n = n0 + t * 16 + (lane & 15);
        float bval = bias[n];
#pragma unroll
        for (int i = 0; i < 8; ++i) {
            out[(size_t)(m0 + mh + i) * EMBED + n] = acc[t][i] + bval;
        }
    }
}

extern "C" void kernel_launch(void* const* d_in, const int* in_sizes, int n_in,
                              void* d_out, int out_size, void* d_ws, size_t ws_size,
                              hipStream_t stream) {
    const float* x    = (const float*)d_in[0];
    const int*   mask = (const int*)d_in[1];
    const float* wq_w = (const float*)d_in[2];
    const float* wq_b = (const float*)d_in[3];
    const float* wk_w = (const float*)d_in[4];
    const float* wk_b = (const float*)d_in[5];
    const float* wv_w = (const float*)d_in[6];
    const float* wv_b = (const float*)d_in[7];
    const float* dw   = (const float*)d_in[8];
    const float* db   = (const float*)d_in[9];

    float* out  = (float*)d_out;                           // [B,S,D]
    float* attn = out + (size_t)BATCH * SEQ * EMBED;       // [B,H,S,S]

    size_t qkvElems = (size_t)BATCH * HEADS * SEQ * DEPTH; // 4 Mi elems
    unsigned short* qb = (unsigned short*)d_ws;
    unsigned short* kb = qb + qkvElems;
    unsigned short* vb = kb + qkvElems;
    float* ctx = (float*)(vb + qkvElems);                  // fp32 [B,S,D]

    dim3 blk(256);
    qkv_kernel<<<dim3(EMBED / 512, (BATCH * SEQ) / 16, 3), blk, 0, stream>>>(
        x, wq_w, wq_b, wk_w, wk_b, wv_w, wv_b, qb, kb, vb);
    logits_kernel<<<dim3(SEQ / 512, SEQ / 16, BATCH * HEADS), blk, 0, stream>>>(
        qb, kb, mask, attn);
    softmax_kernel<<<dim3(BATCH * HEADS * SEQ), blk, 0, stream>>>(attn);
    ctx_kernel<<<dim3(BATCH * HEADS, SEQ / 128), blk, 0, stream>>>(attn, vb, ctx);
    dense_kernel<<<dim3(EMBED / 512, (BATCH * SEQ) / 16), blk, 0, stream>>>(
        ctx, dw, db, out);
}
